// SparsePropConv_6004364280202
// MI455X (gfx1250) — matrix operand order repaired and checked by reading, not compiled, not run on hardware
//
#include <hip/hip_runtime.h>
#include <hip/hip_bf16.h>

// CDNA5 / gfx1250 WMMA types (per bridge doc, codegen-confirmed)
typedef __attribute__((ext_vector_type(16))) _Float16 v16h;
typedef __attribute__((ext_vector_type(8)))  float    v8f;

#define HID 512
#define NCLIP 64
#define BATCH 32
#define NPROP 512
#define XBUF_ELEMS ((size_t)BATCH * NCLIP * HID)   // one activation buffer (b,k,c)

// ---------------------------------------------------------------------------
// Kernel 1: zero map_h (256MB) and set diagonal = x.  float4 stores (B128).
// lin layout: [b:5][c:9][s:6][q:4] -> 16,777,216 threads
// ---------------------------------------------------------------------------
__global__ void spc_init_map(const float* __restrict__ x, float* __restrict__ map_h) {
    unsigned lin = blockIdx.x * blockDim.x + threadIdx.x;
    unsigned q = lin & 15u;
    unsigned s = (lin >> 4) & 63u;
    unsigned c = (lin >> 10) & 511u;
    unsigned b = lin >> 19;
    float xv = x[((size_t)b * HID + c) * NCLIP + s];
    int t0 = (int)(q * 4u);
    float4 v;
    v.x = (t0 + 0 == (int)s) ? xv : 0.0f;
    v.y = (t0 + 1 == (int)s) ? xv : 0.0f;
    v.z = (t0 + 2 == (int)s) ? xv : 0.0f;
    v.w = (t0 + 3 == (int)s) ? xv : 0.0f;
    *(float4*)(map_h + (((size_t)b * HID + c) * NCLIP + s) * NCLIP + t0) = v;
}

// ---------------------------------------------------------------------------
// Kernel 2: map_mask[b,0,s,t] = (0 <= t-s <= 15) ? 1 : 0    (131072 elems)
// ---------------------------------------------------------------------------
__global__ void spc_init_mask(float* __restrict__ mask) {
    unsigned lin = blockIdx.x * blockDim.x + threadIdx.x;
    int t = (int)(lin & 63u);
    int s = (int)((lin >> 6) & 63u);
    int d = t - s;
    mask[lin] = (d >= 0 && d < 16) ? 1.0f : 0.0f;
}

// ---------------------------------------------------------------------------
// Kernel 3: convert w1,w2 (f32, [d][c] row-major) to f16 in workspace
// ---------------------------------------------------------------------------
__global__ void spc_cvt_w(const float* __restrict__ w1, const float* __restrict__ w2,
                          _Float16* __restrict__ w1h, _Float16* __restrict__ w2h) {
    unsigned lin = blockIdx.x * blockDim.x + threadIdx.x;  // 0 .. 2*262144-1
    if (lin < HID * HID) w1h[lin] = (_Float16)w1[lin];
    else                 w2h[lin - HID * HID] = (_Float16)w2[lin - HID * HID];
}

// ---------------------------------------------------------------------------
// Kernel 4: transpose input x (b,c,k) -> xbuf0 (b,k,c) f32.
// lin layout: [b:5][k:6][d:9] -> 1,048,576 threads; stores coalesced.
// ---------------------------------------------------------------------------
__global__ void spc_transpose_x(const float* __restrict__ x, float* __restrict__ xo) {
    unsigned lin = blockIdx.x * blockDim.x + threadIdx.x;
    unsigned d = lin & 511u;
    unsigned k = (lin >> 9) & 63u;
    unsigned b = lin >> 15;
    xo[lin] = x[((size_t)b * HID + d) * NCLIP + k];
}

__device__ __forceinline__ float spc_sigmoid(float v) {
    return 1.0f / (1.0f + __expf(-v));
}

// ---------------------------------------------------------------------------
// Kernel 5: one cross-gate layer, fused dual-GEMM + sigmoid gate + band write.
// Activations live in (b,k,c) layout, c contiguous.
// Each wave owns a 16-row x 32-col tile: A-fragment shared by two B tiles
// -> 4 v_wmma_f32_16x16x32_f16 per K-step.  Rows padded per batch (k<K guard
// on stores; loads clamped to K-1, always in-bounds).
// Tile decomposition is pure shifts: tiles = 32b * 4ktile * 16colpair = 2048.
// ---------------------------------------------------------------------------
__global__ void spc_layer(const float* __restrict__ xcur, float* __restrict__ xnext,
                          const _Float16* __restrict__ w1h, const _Float16* __restrict__ w2h,
                          const float* __restrict__ b1, const float* __restrict__ b2,
                          float* __restrict__ map_h, int Kcols, int layer) {
    const int lane   = threadIdx.x & 31;
    const int waveId = threadIdx.x >> 5;
    const int tileId = blockIdx.x * 8 + waveId;     // 0..2047

    const int tileCol = tileId & 15;                // 16 column pairs (32 cols each)
    const int kTile   = (tileId >> 4) & 3;          // 4 row tiles of 16 per batch
    const int b       = tileId >> 6;                // batch
    const int dBase   = tileCol * 32;

    const int m     = lane & 15;
    const int kHalf = lane >> 4;

    // A-fragment row (clamped; stores are predicated on validity)
    const int kA  = kTile * 16 + m;
    const int kld = (kA < Kcols - 1) ? kA : (Kcols - 1);
    const float* aRow = xcur + ((size_t)b * NCLIP + kld) * HID;   // row kld
    const float* bRow = aRow + HID;                               // row kld+1

    // B-fragments: per lane 16 contiguous f16 (32B aligned)
    const _Float16* w1Row0 = w1h + (size_t)(dBase + m) * HID + 16 * kHalf;
    const _Float16* w1Row1 = w1Row0 + (size_t)16 * HID;
    const _Float16* w2Row0 = w2h + (size_t)(dBase + m) * HID + 16 * kHalf;
    const _Float16* w2Row1 = w2Row0 + (size_t)16 * HID;

    v8f accA0 = {}, accA1 = {};   // a @ w1^T  (cols dBase..+15, dBase+16..+31)
    v8f accB0 = {}, accB1 = {};   // bb @ w2^T

    for (int cBase = 0; cBase < HID; cBase += 32) {
        // A-frag slot e -> c = e + 16*kHalf : one contiguous
        // 16-float chunk -> global_load_b128 pairs + v_cvt_pk_f16_f32
        const float* pa = aRow + cBase + 16 * kHalf;
        const float* pb = bRow + cBase + 16 * kHalf;
        v16h aF, bF;
#pragma unroll
        for (int e = 0; e < 8; ++e) {
            aF[e] = (_Float16)pa[e];
            bF[e] = (_Float16)pb[e];
        }
#pragma unroll
        for (int e = 8; e < 16; ++e) {
            aF[e] = (_Float16)pa[e];
            bF[e] = (_Float16)pb[e];
        }
        v16h w1F0 = *(const v16h*)(w1Row0 + cBase);
        v16h w1F1 = *(const v16h*)(w1Row1 + cBase);
        v16h w2F0 = *(const v16h*)(w2Row0 + cBase);
        v16h w2F1 = *(const v16h*)(w2Row1 + cBase);
        accA0 = __builtin_amdgcn_wmma_f32_16x16x32_f16(false, aF, false, w1F0,
                                                       (short)0, accA0, false, false);
        accA1 = __builtin_amdgcn_wmma_f32_16x16x32_f16(false, aF, false, w1F1,
                                                       (short)0, accA1, false, false);
        accB0 = __builtin_amdgcn_wmma_f32_16x16x32_f16(false, bF, false, w2F0,
                                                       (short)0, accB0, false, false);
        accB1 = __builtin_amdgcn_wmma_f32_16x16x32_f16(false, bF, false, w2F1,
                                                       (short)0, accB1, false, false);
    }

    // C/D layout: vgpr r, lane l -> row M = r + 8*(l>>4), col N = l & 15
    const int d0 = dBase + m;
    const int d1 = d0 + 16;
    const float bias1_0 = b1[d0], bias1_1 = b1[d1];
    const float bias2_0 = b2[d0], bias2_1 = b2[d1];
#pragma unroll
    for (int r = 0; r < 8; ++r) {
        const int ko = kTile * 16 + r + 8 * kHalf;
        if (ko < Kcols) {
            const float* xr = xcur + ((size_t)b * NCLIP + ko) * HID;
            const float a0 = xr[d0],        a1 = xr[d1];
            const float v0 = xr[HID + d0],  v1 = xr[HID + d1];
            const float ga0 = spc_sigmoid(accA0[r] + bias1_0);
            const float ga1 = spc_sigmoid(accA1[r] + bias1_1);
            const float gb0 = spc_sigmoid(accB0[r] + bias2_0);
            const float gb1 = spc_sigmoid(accB1[r] + bias2_1);
            const float o0 = a0 * gb0 + v0 * ga0;
            const float o1 = a1 * gb1 + v1 * ga1;
            map_h[(((size_t)b * HID + d0) * NCLIP + ko) * NCLIP + (ko + layer)] = o0;
            map_h[(((size_t)b * HID + d1) * NCLIP + ko) * NCLIP + (ko + layer)] = o1;
            float* xn = xnext + ((size_t)b * NCLIP + ko) * HID;
            xn[d0] = o0;
            xn[d1] = o1;
        }
    }
}

// ---------------------------------------------------------------------------
// Kernel 6a: props gather from layered activations (fully coalesced).
// x_layers[i] holds layer-i activation (b,k,c); band i == r1-r0.
// ---------------------------------------------------------------------------
__global__ void spc_gather_layers(const float* __restrict__ xlayers,
                                  const long long* __restrict__ props,
                                  float* __restrict__ out) {
    unsigned lin = blockIdx.x * blockDim.x + threadIdx.x;  // 0 .. 8388607
    unsigned d = lin & 511u;
    unsigned p = (lin >> 9) & 511u;
    unsigned b = lin >> 18;
    long long r0 = props[2 * p];
    long long r1 = props[2 * p + 1] - 1;
    long long i  = r1 - r0;                                 // 0..15
    out[lin] = xlayers[(size_t)i * XBUF_ELEMS + ((size_t)b * NCLIP + (size_t)r0) * HID + d];
}

// ---------------------------------------------------------------------------
// Kernel 6b: fallback props gather from map_h (used if ws too small)
// ---------------------------------------------------------------------------
__global__ void spc_gather_map(const float* __restrict__ map_h,
                               const long long* __restrict__ props,
                               float* __restrict__ out) {
    unsigned lin = blockIdx.x * blockDim.x + threadIdx.x;
    unsigned d = lin & 511u;
    unsigned p = (lin >> 9) & 511u;
    unsigned b = lin >> 18;
    long long r0 = props[2 * p];
    long long r1 = props[2 * p + 1] - 1;
    out[lin] = map_h[(((size_t)b * HID + d) * NCLIP + (size_t)r0) * NCLIP + (size_t)r1];
}

// ---------------------------------------------------------------------------
extern "C" void kernel_launch(void* const* d_in, const int* in_sizes, int n_in,
                              void* d_out, int out_size, void* d_ws, size_t ws_size,
                              hipStream_t stream) {
    (void)in_sizes; (void)n_in; (void)out_size;

    const float*     x_in  = (const float*)d_in[0];
    const long long* props = (const long long*)d_in[1];
    // d_in[2] textual_input, d_in[3] textual_mask: unused by reference
    const float* w1 = (const float*)d_in[4];
    const float* b1 = (const float*)d_in[5];
    const float* w2 = (const float*)d_in[6];
    const float* b2 = (const float*)d_in[7];

    float* out = (float*)d_out;
    float* props_h = out;                                     // 32*512*512
    float* map_h   = out + (size_t)BATCH * NPROP * HID;       // +8,388,608
    float* map_msk = map_h + (size_t)BATCH * HID * NCLIP * NCLIP;

    // workspace: f16 weights (1MB) + activation buffers (4MB each)
    _Float16* w1h   = (_Float16*)d_ws;
    _Float16* w2h   = w1h + (size_t)HID * HID;
    float*    xbase = (float*)(w2h + (size_t)HID * HID);

    const size_t needLayered = 2 * (size_t)HID * HID * sizeof(_Float16)
                             + 16 * XBUF_ELEMS * sizeof(float);     // ~65 MB
    const bool layered = (ws_size >= needLayered);

    const int T = 256;

    spc_init_map<<<(BATCH * HID * NCLIP * 16) / T, T, 0, stream>>>(x_in, map_h);
    spc_init_mask<<<(BATCH * NCLIP * NCLIP) / T, T, 0, stream>>>(map_msk);
    spc_cvt_w<<<(2 * HID * HID) / T, T, 0, stream>>>(w1, w2, w1h, w2h);
    spc_transpose_x<<<(BATCH * NCLIP * HID) / T, T, 0, stream>>>(x_in, xbase);

    const int layerBlocks = (BATCH * 4 * (HID / 32)) / 8;  // 2048 tiles / 8 waves
    for (int i = 1; i < 16; ++i) {
        const int Kc = NCLIP - i;
        const float* xcur;
        float*       xnxt;
        if (layered) {
            xcur = xbase + (size_t)(i - 1) * XBUF_ELEMS;
            xnxt = xbase + (size_t)i * XBUF_ELEMS;
        } else {       // ping-pong in two buffers
            xcur = xbase + (size_t)((i + 1) & 1) * XBUF_ELEMS;
            xnxt = xbase + (size_t)(i & 1) * XBUF_ELEMS;
        }
        spc_layer<<<layerBlocks, 256, 0, stream>>>(xcur, xnxt, w1h, w2h,
                                                   b1, b2, map_h, Kc, i);
    }

    if (layered)
        spc_gather_layers<<<(BATCH * NPROP * HID) / T, T, 0, stream>>>(xbase, props, props_h);
    else
        spc_gather_map<<<(BATCH * NPROP * HID) / T, T, 0, stream>>>(map_h, props, props_h);
}
